// VectorQuantiser_34883724378676
// MI455X (gfx1250) — compile-verified, matrix operand order checked by reading, not run on hardware
//
#include <hip/hip_runtime.h>
#include <hip/hip_bf16.h>

typedef __attribute__((ext_vector_type(16))) _Float16 v16h;
typedef __attribute__((ext_vector_type(8)))  _Float16 v8h;
typedef __attribute__((ext_vector_type(8)))  float    v8f;
typedef __attribute__((ext_vector_type(4)))  int      v4i;

#define AS1 __attribute__((address_space(1)))
#define AS3 __attribute__((address_space(3)))

#define N_PIX   65536      // 16*64*64 flattened pixels
#define N_E     1024
#define E_DIM   64
#define ZQ_ELEMS 4194304   // 16*64*64*64

// d_out layout (floats, reference tuple order)
#define OFF_LOSS 4194304
#define OFF_IDX  4194305
#define OFF_SAMP 4259841
#define OFF_MCD  71368705
#define OFF_MCV  71368706

// LDS: 128KB f16 codebook + 4KB norms + pad for harmless last-iteration prefetch over-read
#define LDS_BYTES 136192

__device__ __forceinline__ v16h combine16(v8h lo, v8h hi) {
    return __builtin_shufflevector(lo, hi, 0,1,2,3,4,5,6,7,8,9,10,11,12,13,14,15);
}

// ---------------- init: zero accumulators ----------------
__global__ void k_init(float* p, int n) {
    for (int i = threadIdx.x; i < n; i += blockDim.x) p[i] = 0.0f;
}

// ---------------- prep: f16 codebook copies + norms ----------------
__global__ void k_prep(const float* __restrict__ emb, _Float16* __restrict__ eb,
                       _Float16* __restrict__ ebm2, float* __restrict__ eNorm) {
    int row = blockIdx.x * blockDim.x + threadIdx.x;
    if (row >= N_E) return;
    float s = 0.0f;
    for (int c = 0; c < E_DIM; ++c) {
        float v = emb[row * E_DIM + c];
        s += v * v;
        eb[row * E_DIM + c]   = (_Float16)v;
        ebm2[row * E_DIM + c] = (_Float16)(-2.0f * v);
    }
    eNorm[row] = s;
}

// ---------------- main: fused (-2 z.e + |e|^2) GEMM + argmin via WMMA ----------------
// one wave = one 16-row tile; codebook staged in LDS via async-to-LDS when available;
// B fragments double-buffered against the WMMA pair.
__global__ void __launch_bounds__(256) k_argmin(const float* __restrict__ z,
                                                const _Float16* __restrict__ eb,
                                                const float* __restrict__ eNorm,
                                                int* __restrict__ idxOut,
                                                float* __restrict__ idxF) {
    extern __shared__ char smem[];
    _Float16* ldsEb = (_Float16*)smem;                 // 65536 halves = 128KB
    float*    ldsEn = (float*)(smem + 131072);         // 1024 floats = 4KB (+pad after)

    const int tid  = threadIdx.x;
    const int lane = tid & 31;
    const int wave = tid >> 5;

    // ---- stage f16 codebook into LDS ----
#if defined(__AMDGCN__) && __has_builtin(__builtin_amdgcn_global_load_async_to_lds_b128)
    for (int i = tid; i < 8192; i += 256) {
        __builtin_amdgcn_global_load_async_to_lds_b128(
            (AS1 v4i*)(eb + i * 8), (AS3 v4i*)(ldsEb + i * 8), 0, 0);
    }
  #if __has_builtin(__builtin_amdgcn_s_wait_asynccnt)
    __builtin_amdgcn_s_wait_asynccnt(0);
  #else
    asm volatile("s_wait_asynccnt 0x0" ::: "memory");
  #endif
#else
    for (int i = tid; i < 8192; i += 256)
        ((uint4*)ldsEb)[i] = ((const uint4*)eb)[i];
#endif
    for (int i = tid; i < N_E; i += 256)
        ldsEn[i] = eNorm[i];
    __syncthreads();

    const int rowTile = blockIdx.x * 8 + wave;         // 0..4095
    const int rowbase = rowTile * 16;
    const int mrow    = rowbase + (lane & 15);
    const int kb      = (lane < 16) ? 0 : 8;           // ISA 16-bit A/B layout

    // A fragments: rows of z_flat (transpose gather from [b,c,h,w]), scaled by -2
    const unsigned bimg = (unsigned)mrow >> 12;
    const unsigned hw   = (unsigned)mrow & 4095u;
    const unsigned zbase = (bimg << 18) + hw;          // + (c<<12)
    v16h a0, a1;
    for (int e = 0; e < 8; ++e) {
        a0[e]     = (_Float16)(-2.0f * z[zbase + ((unsigned)(kb +  0 + e) << 12)]);
        a0[e + 8] = (_Float16)(-2.0f * z[zbase + ((unsigned)(kb + 16 + e) << 12)]);
        a1[e]     = (_Float16)(-2.0f * z[zbase + ((unsigned)(kb + 32 + e) << 12)]);
        a1[e + 8] = (_Float16)(-2.0f * z[zbase + ((unsigned)(kb + 48 + e) << 12)]);
    }

    const int col = lane & 15;
    float best[8];
    int   bidx[8];
    for (int r = 0; r < 8; ++r) { best[r] = 3.4e38f; bidx[r] = 0; }

    // double-buffered B fragments (prefetch tile nt+1 while WMMA consumes tile nt)
    const _Float16* r0 = ldsEb + col * E_DIM;
    v8h p0 = *(const v8h*)(r0 + kb);
    v8h p1 = *(const v8h*)(r0 + kb + 16);
    v8h p2 = *(const v8h*)(r0 + kb + 32);
    v8h p3 = *(const v8h*)(r0 + kb + 48);
    float enC = ldsEn[col];

    for (int nt = 0; nt < 64; ++nt) {
        v16h b0 = combine16(p0, p1);
        v16h b1 = combine16(p2, p3);
        const float en = enC;
        // prefetch next tile (last iteration over-reads into LDS pad; never used)
        const _Float16* rn = ldsEb + ((nt + 1) * 16 + col) * E_DIM;
        p0 = *(const v8h*)(rn + kb);
        p1 = *(const v8h*)(rn + kb + 16);
        p2 = *(const v8h*)(rn + kb + 32);
        p3 = *(const v8h*)(rn + kb + 48);
        enC = ldsEn[(nt + 1) * 16 + col];

        v8f acc = {};
        acc = __builtin_amdgcn_wmma_f32_16x16x32_f16(false, a0, false, b0, (short)0, acc, false, false);
        acc = __builtin_amdgcn_wmma_f32_16x16x32_f16(false, a1, false, b1, (short)0, acc, false, false);

        const int code = nt * 16 + col;
        for (int r = 0; r < 8; ++r) {
            float v = acc[r] + en;                     // -2 z.e + |e|^2 (argmin-equivalent dist)
            if (v < best[r]) { best[r] = v; bidx[r] = code; }
        }
    }

    // min-reduce across the 16 columns (stays inside each 16-lane half)
    for (int m = 1; m <= 8; m <<= 1) {
        for (int r = 0; r < 8; ++r) {
            float ov = __shfl_xor(best[r], m, 32);
            int   oi = __shfl_xor(bidx[r], m, 32);
            if (ov < best[r] || (ov == best[r] && oi < bidx[r])) { best[r] = ov; bidx[r] = oi; }
        }
    }
    if ((lane & 15) == 0) {
        const int roff = rowbase + ((lane < 16) ? 0 : 8);  // C layout: lanes 16-31 hold M=8..15
        for (int r = 0; r < 8; ++r) {
            idxOut[roff + r] = bidx[r];
            idxF[roff + r]   = (float)bidx[r];
        }
    }
}

// ---------------- codebook self-distance stats via WMMA ----------------
// one wave = one 16-column strip; loops all 64 row tiles with A-fragment prefetch
__global__ void __launch_bounds__(256) k_cd(const _Float16* __restrict__ ebm2,
                                            const _Float16* __restrict__ eb,
                                            const float* __restrict__ eNorm,
                                            float* __restrict__ rowSum,
                                            float* __restrict__ rowSumSq,
                                            float* __restrict__ min2Sum) {
    const int tid  = threadIdx.x;
    const int lane = tid & 31;
    const int wave = tid >> 5;
    const int jt   = blockIdx.x * 8 + wave;            // 0..63
    const int kb   = (lane < 16) ? 0 : 8;
    const int col  = lane & 15;
    const int j    = jt * 16 + col;

    const _Float16* brow = eb + j * E_DIM;
    v16h b0 = combine16(*(const v8h*)(brow + kb),      *(const v8h*)(brow + kb + 16));
    v16h b1 = combine16(*(const v8h*)(brow + kb + 32), *(const v8h*)(brow + kb + 48));
    const float enj = eNorm[j];
    const int roff  = (lane < 16) ? 0 : 8;

    float m1 = 3.4e38f, m2 = 3.4e38f;                  // two smallest in this column

    // prefetched A fragments
    const _Float16* ar = ebm2 + col * E_DIM;
    v8h q0 = *(const v8h*)(ar + kb);
    v8h q1 = *(const v8h*)(ar + kb + 16);
    v8h q2 = *(const v8h*)(ar + kb + 32);
    v8h q3 = *(const v8h*)(ar + kb + 48);

    for (int it = 0; it < 64; ++it) {
        v16h a0 = combine16(q0, q1);
        v16h a1 = combine16(q2, q3);
        // prefetch next row tile (last iteration over-reads into adjacent ws; never used)
        const _Float16* an = ebm2 + ((it + 1) * 16 + col) * E_DIM;
        q0 = *(const v8h*)(an + kb);
        q1 = *(const v8h*)(an + kb + 16);
        q2 = *(const v8h*)(an + kb + 32);
        q3 = *(const v8h*)(an + kb + 48);

        v8f acc = {};
        acc = __builtin_amdgcn_wmma_f32_16x16x32_f16(false, a0, false, b0, (short)0, acc, false, false);
        acc = __builtin_amdgcn_wmma_f32_16x16x32_f16(false, a1, false, b1, (short)0, acc, false, false);
        for (int r = 0; r < 8; ++r) {
            const int i = it * 16 + roff + r;
            float v = acc[r] + eNorm[i] + enj;         // cd[i][j]
            if (v < m1)      { m2 = m1; m1 = v; }
            else if (v < m2) { m2 = v; }
            // reduce over the 16 columns of this strip for row sums
            float s = v, q = v * v;
            for (int m = 1; m <= 8; m <<= 1) {
                s += __shfl_xor(s, m, 32);
                q += __shfl_xor(q, m, 32);
            }
            if ((lane & 15) == 0) {
                atomicAdd(&rowSum[i], s);
                atomicAdd(&rowSumSq[i], q);
            }
        }
    }
    // merge column halves (rows 0..7 strips live in lane j, rows 8..15 in lane j+16)
    float om1 = __shfl_xor(m1, 16, 32);
    float om2 = __shfl_xor(m2, 16, 32);
    float g2;
    if (m1 <= om1) g2 = fminf(m2, om1);
    else           g2 = fminf(om2, m1);
    if (lane < 16) atomicAdd(min2Sum, g2);             // second-smallest per column
}

// ---------------- z_q gather (exact f32) + loss accumulation ----------------
__global__ void k_out(const float* __restrict__ z, const float* __restrict__ emb,
                      const int* __restrict__ idx, float* __restrict__ zq,
                      float* __restrict__ lossAcc) {
    const unsigned gid = blockIdx.x * 256 + threadIdx.x;   // 4M threads
    const unsigned n = gid & 65535u;
    const unsigned c = gid >> 16;
    const int id = idx[n];
    const float e = emb[id * E_DIM + c];
    const unsigned b = n >> 12, hw = n & 4095u;
    const unsigned pos = (b << 18) + (c << 12) + hw;       // [b,c,h,w] layout
    const float zv = z[pos];
    zq[pos] = e;                                           // straight-through fwd = emb[idx]
    const float d = e - zv;
    __shared__ float red[256];
    red[threadIdx.x] = d * d;
    __syncthreads();
    for (int s = 128; s > 0; s >>= 1) {
        if (threadIdx.x < s) red[threadIdx.x] += red[threadIdx.x + s];
        __syncthreads();
    }
    if (threadIdx.x == 0) atomicAdd(lossAcc, red[0]);
}

// ---------------- sampled_idx: zero fill + scatter ----------------
__global__ void k_zero(float* __restrict__ p, unsigned n) {
    unsigned stride = gridDim.x * blockDim.x;
    for (unsigned i = blockIdx.x * blockDim.x + threadIdx.x; i < n; i += stride)
        p[i] = 0.0f;
}
__global__ void k_scatter(const int* __restrict__ idx, float* __restrict__ samp) {
    unsigned n = blockIdx.x * blockDim.x + threadIdx.x;
    if (n < N_PIX) samp[idx[n]] = 1.0f;                // faithful flat-index semantics
}

// ---------------- finalize scalars ----------------
__global__ void k_final(const float* __restrict__ rowSum, const float* __restrict__ rowSumSq,
                        const float* __restrict__ lossAcc, const float* __restrict__ min2Sum,
                        float* __restrict__ out) {
    __shared__ float red[1024];
    const int t = threadIdx.x;
    float s  = rowSum[t];
    float q  = rowSumSq[t];
    float var = (q - s * s * (1.0f / 1024.0f)) * (1.0f / 1023.0f);  // ddof=1
    red[t] = var;
    __syncthreads();
    for (int st = 512; st > 0; st >>= 1) {
        if (t < st) red[t] += red[t + st];
        __syncthreads();
    }
    if (t == 0) {
        out[OFF_MCV]  = red[0] * (1.0f / 1024.0f);
        out[OFF_MCD]  = min2Sum[0] * (1.0f / 1024.0f);
        out[OFF_LOSS] = 1.2f * lossAcc[0] * (1.0f / (float)ZQ_ELEMS);  // (BETA+1)*MSE
    }
}

extern "C" void kernel_launch(void* const* d_in, const int* in_sizes, int n_in,
                              void* d_out, int out_size, void* d_ws, size_t ws_size,
                              hipStream_t stream) {
    const float* z   = (const float*)d_in[0];
    const float* emb = (const float*)d_in[1];
    float* out = (float*)d_out;

    char* ws = (char*)d_ws;
    _Float16* eb    = (_Float16*)(ws);                 // 128KB
    _Float16* ebm2  = (_Float16*)(ws + 131072);        // 128KB
    float* eNorm    = (float*)(ws + 262144);           // 4KB
    int*   idx      = (int*)(ws + 266240);             // 256KB
    float* rowSum   = (float*)(ws + 528384);           // 4KB
    float* rowSumSq = (float*)(ws + 532480);           // 4KB
    float* lossAcc  = (float*)(ws + 536576);
    float* min2Sum  = (float*)(ws + 536580);

    k_init<<<1, 256, 0, stream>>>(rowSum, 2050);       // rowSum+rowSumSq+lossAcc+min2Sum
    k_prep<<<4, 256, 0, stream>>>(emb, eb, ebm2, eNorm);

    k_argmin<<<512, 256, LDS_BYTES, stream>>>(z, eb, eNorm, idx, out + OFF_IDX);
    k_cd<<<8, 256, 0, stream>>>(ebm2, eb, eNorm, rowSum, rowSumSq, min2Sum);

    k_zero<<<8192, 256, 0, stream>>>(out + OFF_SAMP, 67108864u);
    k_scatter<<<256, 256, 0, stream>>>(idx, out + OFF_SAMP);

    k_out<<<16384, 256, 0, stream>>>(z, emb, idx, out /*zq at offset 0*/, lossAcc);
    k_final<<<1, 1024, 0, stream>>>(rowSum, rowSumSq, lossAcc, min2Sum, out);
}
